// CrossAttention_58171037057362
// MI455X (gfx1250) — matrix operand order repaired and checked by reading, not compiled, not run on hardware
//
#include <hip/hip_runtime.h>
#include <hip/hip_bf16.h>

#define EMBED   1024
#define NHEADS  16
#define HDIM    64
#define BATCH   2
#define SEQ     2048
#define MTOT    (BATCH*SEQ)   // 4096 rows for the big GEMMs

typedef __bf16 bf16;
typedef __attribute__((ext_vector_type(16))) __bf16 v16bf;
typedef __attribute__((ext_vector_type(8)))  __bf16 v8bf;
typedef __attribute__((ext_vector_type(8)))  float  v8f;
typedef int v4i __attribute__((vector_size(16)));   // matches builtin param type

#define WMMA_BF16(A,B,C) \
  __builtin_amdgcn_wmma_f32_16x16x32_bf16(false,(A),false,(B),(short)0,(C),false,false)

// ---- gfx1250 async global->LDS path (builtin confirmed present; param types
// from the round-3 diagnostic: (v4i as1*, v4i as3*, imm, imm)) ----
#if defined(__has_builtin)
#  if __has_builtin(__builtin_amdgcn_global_load_async_to_lds_b128)
#    define HAVE_ASYNC_LDS 1
#  endif
#endif
#ifndef HAVE_ASYNC_LDS
#  define HAVE_ASYNC_LDS 0
#endif

#if HAVE_ASYNC_LDS
#define ASYNC_COPY16(g, l)                                                       \
  __builtin_amdgcn_global_load_async_to_lds_b128(                                \
      (__attribute__((address_space(1))) v4i*)(unsigned long long)(g),           \
      (__attribute__((address_space(3))) v4i*)(unsigned)(unsigned long long)(l), \
      0, 0)
#if __has_builtin(__builtin_amdgcn_s_wait_asynccnt)
#define ASYNC_WAIT() __builtin_amdgcn_s_wait_asynccnt(0)
#else
#define ASYNC_WAIT() asm volatile("s_wait_asynccnt 0x0" ::: "memory")
#endif
#else
#define ASYNC_COPY16(g, l) (*(v8bf*)(l) = *(const v8bf*)(g))
#define ASYNC_WAIT() ((void)0)
#endif

static __device__ __forceinline__ v16bf wmma_pack(v8bf lo, v8bf hi) {
  v16bf r;
#pragma unroll
  for (int e = 0; e < 8; ++e) { r[e] = lo[e]; r[e + 8] = hi[e]; }
  return r;
}

// A fragment (16x32 bf16, row = lane&15): elem e = k[kb + e]
// where kb = (lane>>4)*16 is folded into the pointer by the caller.
static __device__ __forceinline__ v16bf load_a_bf(const bf16* p) {
  return wmma_pack(*(const v8bf*)p, *(const v8bf*)(p + 8));
}
// B fragment (32x16 bf16, col = lane&15): elem e = k[(lane>>4)*16 + e], pointer pre-offset.
static __device__ __forceinline__ v16bf load_b_bf(const bf16* p) {
  return wmma_pack(*(const v8bf*)p, *(const v8bf*)(p + 8));
}

// ---------------------------------------------------------------------------
// Bulk fp32 -> bf16 conversion (8 elements/thread, b128 in / b128 out).
// ---------------------------------------------------------------------------
__global__ __launch_bounds__(256)
void cvt_bf16_kernel(const float* __restrict__ src, bf16* __restrict__ dst, int n)
{
  const int i = (blockIdx.x * 256 + threadIdx.x) * 8;
  if (i >= n) return;
  const float4 x = *(const float4*)(src + i);
  const float4 y = *(const float4*)(src + i + 4);
  v8bf r;
  r[0] = (bf16)x.x; r[1] = (bf16)x.y; r[2] = (bf16)x.z; r[3] = (bf16)x.w;
  r[4] = (bf16)y.x; r[5] = (bf16)y.y; r[6] = (bf16)y.z; r[7] = (bf16)y.w;
  *(v8bf*)(dst + i) = r;
}

// ---------------------------------------------------------------------------
// Projection GEMM (all-bf16 operands): dst = X @ W^T + bias, scattered to the
// attention layout. vmode==0 -> [B,H,T,D]; vmode==1 -> V transposed [B,H,D,S].
// Block = 8 waves; wave computes 32(M) x 64(N) with 8 WMMAs per 32-wide k-step.
// ---------------------------------------------------------------------------
__global__ __launch_bounds__(256)
void proj_gemm_kernel(const bf16* __restrict__ X, const bf16* __restrict__ W,
                      const float* __restrict__ bias, bf16* __restrict__ dst, int vmode)
{
  const int lane  = threadIdx.x & 31;
  const int wave  = threadIdx.x >> 5;
  const int l16   = lane & 15;
  const int lh    = lane >> 4;
  const int mbase = blockIdx.y * 256 + wave * 32;
  const int nbase = blockIdx.x * 64;

  v8f acc[8] = {};
  const bf16* ar0 = X + (size_t)(mbase +      l16) * EMBED + lh * 16;
  const bf16* ar1 = X + (size_t)(mbase + 16 + l16) * EMBED + lh * 16;
  const bf16* wr0 = W + (size_t)(nbase +  0 + l16) * EMBED + lh * 16;
  const bf16* wr1 = W + (size_t)(nbase + 16 + l16) * EMBED + lh * 16;
  const bf16* wr2 = W + (size_t)(nbase + 32 + l16) * EMBED + lh * 16;
  const bf16* wr3 = W + (size_t)(nbase + 48 + l16) * EMBED + lh * 16;

  for (int kt = 0; kt < EMBED; kt += 32) {
    const v16bf a0 = load_a_bf(ar0 + kt);
    const v16bf a1 = load_a_bf(ar1 + kt);
    const v16bf b0 = load_b_bf(wr0 + kt);
    const v16bf b1 = load_b_bf(wr1 + kt);
    const v16bf b2 = load_b_bf(wr2 + kt);
    const v16bf b3 = load_b_bf(wr3 + kt);
    acc[0] = WMMA_BF16(a0, b0, acc[0]);
    acc[1] = WMMA_BF16(a0, b1, acc[1]);
    acc[2] = WMMA_BF16(a0, b2, acc[2]);
    acc[3] = WMMA_BF16(a0, b3, acc[3]);
    acc[4] = WMMA_BF16(a1, b0, acc[4]);
    acc[5] = WMMA_BF16(a1, b1, acc[5]);
    acc[6] = WMMA_BF16(a1, b2, acc[6]);
    acc[7] = WMMA_BF16(a1, b3, acc[7]);
  }

#pragma unroll
  for (int mt = 0; mt < 2; ++mt) {
#pragma unroll
    for (int nt = 0; nt < 4; ++nt) {
      const int n = nbase + nt * 16 + l16;
      const float bb = bias[n];
      const int h = n >> 6, d = n & (HDIM - 1);
#pragma unroll
      for (int v = 0; v < 8; ++v) {
        const int m = mbase + mt * 16 + v + lh * 8;
        const int b = m >> 11, t = m & (SEQ - 1);
        const float val = acc[mt * 4 + nt][v] + bb;
        size_t off;
        if (vmode) off = (((size_t)b * NHEADS + h) * HDIM + d) * SEQ + t;  // V^T [B,H,D,S]
        else       off = (((size_t)b * NHEADS + h) * SEQ + t) * HDIM + d;  // [B,H,T,D]
        dst[off] = (bf16)val;
      }
    }
  }
}

// ---------------------------------------------------------------------------
// Output GEMM: out = A(bf16) @ Wo^T(bf16) + bo, fp32 row-major output.
// ---------------------------------------------------------------------------
__global__ __launch_bounds__(256)
void out_gemm_kernel(const bf16* __restrict__ A, const bf16* __restrict__ W,
                     const float* __restrict__ bias, float* __restrict__ out)
{
  const int lane  = threadIdx.x & 31;
  const int wave  = threadIdx.x >> 5;
  const int l16   = lane & 15;
  const int lh    = lane >> 4;
  const int mbase = blockIdx.y * 256 + wave * 32;
  const int nbase = blockIdx.x * 64;

  v8f acc[8] = {};
  const bf16* ar0 = A + (size_t)(mbase +      l16) * EMBED + lh * 16;
  const bf16* ar1 = A + (size_t)(mbase + 16 + l16) * EMBED + lh * 16;
  const bf16* wr0 = W + (size_t)(nbase +  0 + l16) * EMBED + lh * 16;
  const bf16* wr1 = W + (size_t)(nbase + 16 + l16) * EMBED + lh * 16;
  const bf16* wr2 = W + (size_t)(nbase + 32 + l16) * EMBED + lh * 16;
  const bf16* wr3 = W + (size_t)(nbase + 48 + l16) * EMBED + lh * 16;

  for (int kt = 0; kt < EMBED; kt += 32) {
    const v16bf a0 = load_a_bf(ar0 + kt);
    const v16bf a1 = load_a_bf(ar1 + kt);
    const v16bf b0 = load_b_bf(wr0 + kt);
    const v16bf b1 = load_b_bf(wr1 + kt);
    const v16bf b2 = load_b_bf(wr2 + kt);
    const v16bf b3 = load_b_bf(wr3 + kt);
    acc[0] = WMMA_BF16(a0, b0, acc[0]);
    acc[1] = WMMA_BF16(a0, b1, acc[1]);
    acc[2] = WMMA_BF16(a0, b2, acc[2]);
    acc[3] = WMMA_BF16(a0, b3, acc[3]);
    acc[4] = WMMA_BF16(a1, b0, acc[4]);
    acc[5] = WMMA_BF16(a1, b1, acc[5]);
    acc[6] = WMMA_BF16(a1, b2, acc[6]);
    acc[7] = WMMA_BF16(a1, b3, acc[7]);
  }

#pragma unroll
  for (int mt = 0; mt < 2; ++mt) {
#pragma unroll
    for (int nt = 0; nt < 4; ++nt) {
      const int n = nbase + nt * 16 + l16;
      const float bb = bias[n];
#pragma unroll
      for (int v = 0; v < 8; ++v) {
        const int m = mbase + mt * 16 + v + lh * 8;
        out[(size_t)m * EMBED + n] = acc[mt * 4 + nt][v] + bb;
      }
    }
  }
}

// ---------------------------------------------------------------------------
// Causal flash attention, wave32 WMMA, block-cooperative K/V staging.
// Block = 4 waves = 64 query rows of one (b,h). All waves run the same s-loop
// to the block causal bound; K (32x64) and V^T (64x32) tiles are staged once
// per block into double-buffered LDS with async global->LDS loads, then each
// wave computes from LDS. Per-wave work is predicated wave-uniformly (EXEC
// stays all-1s for WMMA).
// ---------------------------------------------------------------------------
__global__ __launch_bounds__(128)
void flash_attn_kernel(const bf16* __restrict__ Q, const bf16* __restrict__ K,
                       const bf16* __restrict__ Vt, const unsigned char* __restrict__ pad,
                       bf16* __restrict__ O)
{
  __shared__ __align__(16) bf16 kbuf[2][32][64];   // [buf][s_local][d]
  __shared__ __align__(16) bf16 vbuf[2][64][32];   // [buf][d][s_local]
  __shared__ __align__(16) bf16 plds[4][16][32];   // per-wave P restage

  const int tid  = threadIdx.x;
  const int lane = tid & 31, wave = tid >> 5;
  const int l16 = lane & 15, lh = lane >> 4;
  const int bh = blockIdx.y, b = bh >> 4, h = bh & (NHEADS - 1);
  const int mblk  = blockIdx.x * 64;
  const int mbase = mblk + wave * 16;
  const int smax  = mblk + 64;        // block causal bound (uniform across block)
  const int send  = mbase + 16;       // wave causal bound (wave-uniform)

  const bf16* q  = Q  + (size_t)bh * SEQ * HDIM;
  const bf16* kk = K  + (size_t)bh * SEQ * HDIM;
  const bf16* vt = Vt + (size_t)bh * HDIM * SEQ;
  const unsigned char* pm = pad + (size_t)b * SEQ;

  // Cooperative stage of one 32-key step: K tile 32x64, V^T tile 64x32
  // (256 16-byte chunks each; 2 per thread per tile).
  auto stage = [&](int buf, int s0) {
#pragma unroll
    for (int i = 0; i < 2; ++i) {
      const int c = tid * 2 + i;
      const int kr = c >> 3, kc = (c & 7) * 8;
      ASYNC_COPY16(kk + (size_t)(s0 + kr) * HDIM + kc, &kbuf[buf][kr][kc]);
      const int vr = c >> 2, vc = (c & 3) * 8;
      ASYNC_COPY16(vt + (size_t)vr * SEQ + s0 + vc, &vbuf[buf][vr][vc]);
    }
  };

  const bf16* qrow = q + (size_t)(mbase + l16) * HDIM;
  const v16bf qa0 = load_a_bf(qrow + lh * 16);        // d 0..31 tile
  const v16bf qa1 = load_a_bf(qrow + 32 + lh * 16);   // d 32..63 tile

  float mrow[8], lrow[8];
  v8f oacc[4] = {};
#pragma unroll
  for (int v = 0; v < 8; ++v) { mrow[v] = -3.0e38f; lrow[v] = 0.f; }

  const float scale = 0.125f;      // HEAD_DIM^-0.5
  const int nIter = smax / 32;

  stage(0, 0);
  for (int it = 0; it < nIter; ++it) {
    const int s0 = it * 32;
    const int buf = it & 1;
    ASYNC_WAIT();
    __syncthreads();                       // staged tile visible to all waves
    if (it + 1 < nIter) stage(buf ^ 1, s0 + 32);

    if (s0 < send) {                       // wave-uniform predicate
      // ---- scores: two 16x16 tiles over d = 0..63, K from LDS ----
      v8f sc[2];
#pragma unroll
      for (int st = 0; st < 2; ++st) {
        const bf16* krow = &kbuf[buf][st * 16 + l16][0];
        v8f t = {};
        t = WMMA_BF16(qa0, load_b_bf(krow + lh * 16), t);
        t = WMMA_BF16(qa1, load_b_bf(krow + 32 + lh * 16), t);
        sc[st] = t;
      }
      // ---- scale + causal + padding mask ----
#pragma unroll
      for (int st = 0; st < 2; ++st) {
        const int s = s0 + st * 16 + l16;
        const bool dead = pm[s] != 0;
#pragma unroll
        for (int v = 0; v < 8; ++v) {
          const int t_idx = mbase + v + lh * 8;
          float x = sc[st][v] * scale;
          if (s > t_idx || dead) x = -3.0e38f;
          sc[st][v] = x;
        }
      }
      // ---- online softmax ----
      float cmax[8];
#pragma unroll
      for (int v = 0; v < 8; ++v) cmax[v] = fmaxf(sc[0][v], sc[1][v]);
#pragma unroll
      for (int dd = 1; dd < 16; dd <<= 1)
#pragma unroll
        for (int v = 0; v < 8; ++v) cmax[v] = fmaxf(cmax[v], __shfl_xor(cmax[v], dd, 32));
      float alpha[8];
#pragma unroll
      for (int v = 0; v < 8; ++v) {
        const float nm = fmaxf(mrow[v], cmax[v]);
        alpha[v] = __expf(mrow[v] - nm);
        mrow[v] = nm;
      }
#pragma unroll
      for (int st = 0; st < 2; ++st)
#pragma unroll
        for (int v = 0; v < 8; ++v) sc[st][v] = __expf(sc[st][v] - mrow[v]);
      float rsum[8];
#pragma unroll
      for (int v = 0; v < 8; ++v) rsum[v] = sc[0][v] + sc[1][v];
#pragma unroll
      for (int dd = 1; dd < 16; dd <<= 1)
#pragma unroll
        for (int v = 0; v < 8; ++v) rsum[v] += __shfl_xor(rsum[v], dd, 32);
#pragma unroll
      for (int v = 0; v < 8; ++v) lrow[v] = lrow[v] * alpha[v] + rsum[v];
#pragma unroll
      for (int nt = 0; nt < 4; ++nt)
#pragma unroll
        for (int v = 0; v < 8; ++v) oacc[nt][v] *= alpha[v];

      // ---- restage P (16x32 bf16) through LDS into A-fragment layout ----
#pragma unroll
      for (int st = 0; st < 2; ++st)
#pragma unroll
        for (int v = 0; v < 8; ++v)
          plds[wave][v + lh * 8][st * 16 + l16] = (bf16)sc[st][v];
      asm volatile("s_wait_dscnt 0" ::: "memory");
      const v16bf pa = load_a_bf(&plds[wave][l16][lh * 16]);

      // ---- O += P @ V, V^T from LDS ----
#pragma unroll
      for (int nt = 0; nt < 4; ++nt) {
        const bf16* vrow = &vbuf[buf][nt * 16 + l16][lh * 16];
        oacc[nt] = WMMA_BF16(pa, load_b_bf(vrow), oacc[nt]);
      }
    }
    __syncthreads();                       // all reads of buf done before restage
  }

  // ---- normalize and store to [B, T, E] bf16 ----
#pragma unroll
  for (int nt = 0; nt < 4; ++nt) {
    const int d = nt * 16 + l16;
#pragma unroll
    for (int v = 0; v < 8; ++v) {
      const int t_idx = mbase + v + lh * 8;
      const float val = oacc[nt][v] / lrow[v];
      O[((size_t)b * SEQ + t_idx) * EMBED + (size_t)h * HDIM + d] = (bf16)val;
    }
  }
}

// ---------------------------------------------------------------------------
extern "C" void kernel_launch(void* const* d_in, const int* in_sizes, int n_in,
                              void* d_out, int out_size, void* d_ws, size_t ws_size,
                              hipStream_t stream) {
  const float* query = (const float*)d_in[0];
  const float* key   = (const float*)d_in[1];
  const float* value = (const float*)d_in[2];
  const unsigned char* mask = (const unsigned char*)d_in[3];
  const float* Wq = (const float*)d_in[4];  const float* bq = (const float*)d_in[5];
  const float* Wk = (const float*)d_in[6];  const float* bk = (const float*)d_in[7];
  const float* Wv = (const float*)d_in[8];  const float* bv = (const float*)d_in[9];
  const float* Wo = (const float*)d_in[10]; const float* bo = (const float*)d_in[11];
  float* out = (float*)d_out;

  const size_t act = (size_t)MTOT * EMBED;   // 4M elements
  const size_t wgt = (size_t)EMBED * EMBED;  // 1M elements
  bf16* p = (bf16*)d_ws;
  bf16* xq  = p; p += act;   // bf16 query
  bf16* xk  = p; p += act;   // bf16 key
  bf16* xv  = p; p += act;   // bf16 value
  bf16* wq  = p; p += wgt;   // bf16 weights
  bf16* wk  = p; p += wgt;
  bf16* wv  = p; p += wgt;
  bf16* wo  = p; p += wgt;
  bf16* qb  = p; p += act;   // Q [B,H,T,D]
  bf16* kb  = p; p += act;   // K [B,H,T,D]
  bf16* vtb = p; p += act;   // V^T [B,H,D,S]
  bf16* ab  = p; p += act;   // attention output [B,T,E]

  const int cvt_blocks_act = (int)(act / 8 / 256);
  const int cvt_blocks_wgt = (int)(wgt / 8 / 256);
  cvt_bf16_kernel<<<cvt_blocks_act, 256, 0, stream>>>(query, xq, (int)act);
  cvt_bf16_kernel<<<cvt_blocks_act, 256, 0, stream>>>(key,   xk, (int)act);
  cvt_bf16_kernel<<<cvt_blocks_act, 256, 0, stream>>>(value, xv, (int)act);
  cvt_bf16_kernel<<<cvt_blocks_wgt, 256, 0, stream>>>(Wq, wq, (int)wgt);
  cvt_bf16_kernel<<<cvt_blocks_wgt, 256, 0, stream>>>(Wk, wk, (int)wgt);
  cvt_bf16_kernel<<<cvt_blocks_wgt, 256, 0, stream>>>(Wv, wv, (int)wgt);
  cvt_bf16_kernel<<<cvt_blocks_wgt, 256, 0, stream>>>(Wo, wo, (int)wgt);

  const dim3 gg(EMBED / 64, MTOT / 256);   // 16 x 16 blocks, 256 threads
  proj_gemm_kernel<<<gg, 256, 0, stream>>>(xq, wq, bq, qb, 0);
  proj_gemm_kernel<<<gg, 256, 0, stream>>>(xk, wk, bk, kb, 0);
  proj_gemm_kernel<<<gg, 256, 0, stream>>>(xv, wv, bv, vtb, 1);

  flash_attn_kernel<<<dim3(SEQ / 64, BATCH * NHEADS), 128, 0, stream>>>(qb, kb, vtb, mask, ab);

  out_gemm_kernel<<<gg, 256, 0, stream>>>(ab, wo, bo, out);
}
